// HSTUBlock_35175782154742
// MI455X (gfx1250) — matrix operand order repaired and checked by reading, not compiled, not run on hardware
//
#include <hip/hip_runtime.h>
#include <hip/hip_bf16.h>

typedef __attribute__((ext_vector_type(16))) _Float16 v16h;
typedef __attribute__((ext_vector_type(8)))  _Float16 v8h;
typedef __attribute__((ext_vector_type(8)))  float    v8f;
typedef __attribute__((ext_vector_type(4)))  unsigned u32x4;
typedef __attribute__((ext_vector_type(4)))  int      i32x4;
typedef __attribute__((ext_vector_type(8)))  int      i32x8;

#define BATCH  2
#define SEQ    2048
#define HID    1024
#define FFN    4096
#define HEADS  16
#define HD     64
#define MAXSEQ 2048

#if __has_builtin(__builtin_amdgcn_tensor_load_to_lds)
#define HAVE_TDM 1
#else
#define HAVE_TDM 0
#endif

// fast SiLU: single v_exp + v_rcp (activation precision is ample)
__device__ __forceinline__ float silu_f(float x) {
  return x * __builtin_amdgcn_rcpf(1.0f + __expf(-x));
}

// ---- CDNA5 async global->LDS copy (ASYNCcnt-tracked, 16B per lane) ----
__device__ __forceinline__ void async_copy_b128(const _Float16* g, _Float16* lds) {
  unsigned l = (unsigned)(unsigned long long)(const void*)lds;
  asm volatile("global_load_async_to_lds_b128 %0, %1, off"
               :: "v"(l), "v"(g) : "memory");
}
__device__ __forceinline__ void wait_async0() {
  asm volatile("s_wait_asynccnt 0x0" ::: "memory");
}

#if HAVE_TDM
// ---- Tensor Data Mover: 2D f16 tile, global(strided) -> LDS(padded rows) ----
// D# layout per CDNA5 ISA. rows/rowlen/stride in
// elements (data_size = 2B). padIC: log2(row_bytes/8); padAC: pad_dwords-1.
__device__ __forceinline__ void tdm_load_2d(const _Float16* g, const _Float16* lds,
                                            unsigned rows, unsigned rowlen,
                                            unsigned stride,
                                            unsigned padIC, unsigned padAC) {
  unsigned long long ga = (unsigned long long)g;
  unsigned la = (unsigned)(unsigned long long)(const void*)lds;
  u32x4 g0;
  g0[0] = 1u;                                     // count=1, user mode
  g0[1] = la;                                     // lds_addr (bytes)
  g0[2] = (unsigned)ga;                           // global_addr[31:0]
  g0[3] = (unsigned)(ga >> 32) | (2u << 30);      // global_addr[56:32] | type=2
  i32x8 g1;
  g1[0] = (int)((1u << 16)                        // data_size = 2 bytes
              | (1u << 20)                        // pad_enable
              | (padIC << 22) | (padAC << 25));
  g1[1] = (int)((rowlen & 0xFFFFu) << 16);        // tensor_dim0[15:0]
  g1[2] = (int)((rowlen >> 16) | ((rows & 0xFFFFu) << 16));   // dim0 hi | dim1 lo
  g1[3] = (int)((rows >> 16) | (rowlen << 16));   // dim1 hi | tile_dim0
  g1[4] = (int)(rows & 0xFFFFu);                  // tile_dim1 | tile_dim2=0
  g1[5] = (int)stride;                            // tensor_dim0_stride[31:0]
  g1[6] = 0;                                      // stride hi | dim1_stride lo
  g1[7] = 0;
  i32x4 z4 = {0, 0, 0, 0};
#if defined(__clang_major__) && __clang_major__ >= 23
  i32x8 z8 = {0, 0, 0, 0, 0, 0, 0, 0};
  __builtin_amdgcn_tensor_load_to_lds(g0, g1, z4, z4, z8, 0);
#else
  __builtin_amdgcn_tensor_load_to_lds(g0, g1, z4, z4, 0);
#endif
}
#endif

// ---- WMMA fragment loaders (CDNA5 wave32 layouts, CDNA5 ISA) ----
__device__ __forceinline__ v16h lds_a_frag(const _Float16* base, int ld, int row0, int col0) {
  const unsigned lane = threadIdx.x & 31u;
  const _Float16* p = base + (size_t)(row0 + (lane & 15u)) * ld + col0 + ((lane >> 4) << 3);
  union { v16h v; unsigned u[8]; } r;
#pragma unroll
  for (int i = 0; i < 4; ++i) r.u[i]     = *(const unsigned*)(p + 2 * i);
#pragma unroll
  for (int i = 0; i < 4; ++i) r.u[4 + i] = *(const unsigned*)(p + 16 + 2 * i);
  return r.v;
}

__device__ __forceinline__ v16h lds_b_frag(const _Float16* base, int ld, int krow0, int col0) {
  const unsigned lane = threadIdx.x & 31u;
  const _Float16* p = base + (size_t)(krow0 + ((lane >> 4) << 3)) * ld + col0 + (lane & 15u);
  v16h r;
#pragma unroll
  for (int i = 0; i < 8; ++i) r[i]     = p[(size_t)i * ld];
#pragma unroll
  for (int i = 0; i < 8; ++i) r[8 + i] = p[(size_t)(16 + i) * ld];
  return r;
}

__device__ __forceinline__ v8f wmma_f16(v16h a, v16h b, v8f c) {
  return __builtin_amdgcn_wmma_f32_16x16x32_f16(false, a, false, b, (short)0, c, false, false);
}

#define C_ROW(i, lane) ((i) + (((lane) >> 4) << 3))
#define C_COL(lane)    ((lane) & 15)

// =====================================================================
// Kernel 0: one-pass fp32 -> f16 conversion of x, W1, W2.
// =====================================================================
__global__ __launch_bounds__(256)
void k_cvt(const float* __restrict__ x, const float* __restrict__ W1,
           const float* __restrict__ W2,
           _Float16* __restrict__ xh, _Float16* __restrict__ W1h,
           _Float16* __restrict__ W2h) {
  const size_t NX  = (size_t)BATCH * SEQ * HID;
  const size_t NW1 = (size_t)HID * FFN;
  const size_t NW2 = (size_t)HID * HID;
  size_t i = ((size_t)blockIdx.x * 256 + threadIdx.x) * 8;
  const float* src; _Float16* dst; size_t off;
  if (i < NX)                 { src = x;  dst = xh;  off = i; }
  else if (i < NX + NW1)      { src = W1; dst = W1h; off = i - NX; }
  else if (i < NX + NW1+NW2)  { src = W2; dst = W2h; off = i - NX - NW1; }
  else return;
  float4 a = *(const float4*)(src + off);
  float4 b = *(const float4*)(src + off + 4);
  v8h h = { (_Float16)a.x, (_Float16)a.y, (_Float16)a.z, (_Float16)a.w,
            (_Float16)b.x, (_Float16)b.y, (_Float16)b.z, (_Float16)b.w };
  *(v8h*)(dst + off) = h;
}

// ---- per-lane async tile fills for the 128x128 GEMM kernels ----
__device__ __forceinline__ void fill_gemm_tiles(const _Float16* __restrict__ Ag, int ldA,
                                                const _Float16* __restrict__ Bg, int ldB,
                                                _Float16* As, _Float16* Bs,
                                                int tid, int m0, int n0, int k0) {
#pragma unroll
  for (int e = 0; e < 2; ++e) {          // A: 128x32 -> 512 chunks of 8 halves
    int c = e * 256 + tid;
    int r = c >> 2, col = (c & 3) * 8;
    async_copy_b128(Ag + (size_t)(m0 + r) * ldA + k0 + col, As + r * 40 + col);
  }
#pragma unroll
  for (int e = 0; e < 2; ++e) {          // B: 32x128 -> 512 chunks
    int c = e * 256 + tid;
    int r = c >> 4, col = (c & 15) * 8;
    async_copy_b128(Bg + (size_t)(k0 + r) * ldB + n0 + col, Bs + r * 136 + col);
  }
}

// =====================================================================
// Kernel 1: activated = silu(xh @ W1h + b1); scatter U(f16), Q[b,h,s,d],
// K[b,h,d,s] (pre-transposed), V[b,h,s,d]. Double-buffered async fills.
// =====================================================================
__global__ __launch_bounds__(256)
void k_proj(const _Float16* __restrict__ xh, const _Float16* __restrict__ W1h,
            const float* __restrict__ b1,
            _Float16* __restrict__ Uh, _Float16* __restrict__ Qh,
            _Float16* __restrict__ Ktg, _Float16* __restrict__ Vh) {
  __shared__ __align__(16) _Float16 As[2][128 * 40];
  __shared__ __align__(16) _Float16 Bs[2][32 * 136];

  const int m0 = blockIdx.x * 128;
  const int n0 = blockIdx.y * 128;
  const int tid = threadIdx.x;
  const int w = tid >> 5;
  const unsigned lane = tid & 31u;
  const int wm = (w & 3) * 32;
  const int wn = (w >> 2) * 64;

  const v8f vzero = {0.f, 0.f, 0.f, 0.f, 0.f, 0.f, 0.f, 0.f};
  v8f acc[2][4];
#pragma unroll
  for (int i = 0; i < 2; ++i)
#pragma unroll
    for (int j = 0; j < 4; ++j) acc[i][j] = vzero;

  fill_gemm_tiles(xh, HID, W1h, FFN, As[0], Bs[0], tid, m0, n0, 0);

  for (int it = 0; it < HID / 32; ++it) {
    wait_async0();
    __syncthreads();
    const int cur = it & 1;
    if (it + 1 < HID / 32)
      fill_gemm_tiles(xh, HID, W1h, FFN, As[1 - cur], Bs[1 - cur], tid, m0, n0, (it + 1) * 32);

    v16h a[2], b[4];
#pragma unroll
    for (int i = 0; i < 2; ++i) a[i] = lds_a_frag(As[cur], 40, wm + i * 16, 0);
#pragma unroll
    for (int j = 0; j < 4; ++j) b[j] = lds_b_frag(Bs[cur], 136, 0, wn + j * 16);
#pragma unroll
    for (int i = 0; i < 2; ++i)
#pragma unroll
      for (int j = 0; j < 4; ++j)
        acc[i][j] = wmma_f16(a[i], b[j], acc[i][j]);
  }

#pragma unroll
  for (int i = 0; i < 2; ++i)
#pragma unroll
    for (int j = 0; j < 4; ++j)
#pragma unroll
      for (int r = 0; r < 8; ++r) {
        int row = m0 + wm + i * 16 + C_ROW(r, lane);
        int col = n0 + wn + j * 16 + C_COL(lane);
        float v = silu_f(acc[i][j][r] + b1[col]);
        _Float16 hv = (_Float16)v;
        if (col < HID) {
          Uh[(size_t)row * HID + col] = hv;
        } else {
          int bb = row >> 11, s = row & (SEQ - 1);
          int c = col & (HID - 1);
          int which = col >> 10;             // 1=Q, 2=K, 3=V
          int head = c >> 6, d = c & 63;
          size_t bh = (size_t)(bb * HEADS + head);
          if (which == 1)      Qh [(bh * SEQ + s) * HD + d] = hv;
          else if (which == 2) Ktg[(bh * HD + d) * SEQ + s] = hv;  // [b,h,d,s]
          else                 Vh [(bh * SEQ + s) * HD + d] = hv;
        }
      }
}

// =====================================================================
// Kernel 2: causal SiLU attention + rel-pos bias, fused with U gate.
// Q/K/V tiles moved by the Tensor Data Mover (wave 0 issues the DMA,
// LDS row padding 64->72 halves done by TDM pad fields).
// =====================================================================
__global__ __launch_bounds__(256)
void k_attn(const _Float16* __restrict__ Qh, const _Float16* __restrict__ Ktg,
            const _Float16* __restrict__ Vh, const _Float16* __restrict__ Uh,
            const float* __restrict__ rel_table, _Float16* __restrict__ G) {
  __shared__ __align__(16) _Float16 Qs[64 * 72];  // [q][d]
  __shared__ __align__(16) _Float16 Kt[64 * 72];  // [d][k]
  __shared__ __align__(16) _Float16 Vs[64 * 72];  // [k][d]
  __shared__ __align__(16) _Float16 Ss[64 * 72];  // [q][k] post-activation

  const int bh = blockIdx.x;            // b*HEADS + h
  const int h  = bh & (HEADS - 1);
  const int bb = bh >> 4;
  const int qt = blockIdx.y;
  const int q0 = qt * 64;
  const int tid = threadIdx.x;
  const int w = tid >> 5;
  const unsigned lane = tid & 31u;
  const float scale = 0.125f;           // HD^-0.5

#if HAVE_TDM
  if (w == 0)                           // one DMA for the resident Q tile
    tdm_load_2d(Qh + ((size_t)bh * SEQ + q0) * HD, Qs, 64, 64, HD, 4, 3);
#else
#pragma unroll
  for (int e = 0; e < 2; ++e) {
    int c = e * 256 + tid;
    int r = c >> 3, col = (c & 7) * 8;
    async_copy_b128(Qh + ((size_t)bh * SEQ + q0 + r) * HD + col, Qs + r * 72 + col);
  }
#endif

  const int sqr = (w & 3) * 16;
  const int skc = (w >> 2) * 32;
  const v8f vzero = {0.f, 0.f, 0.f, 0.f, 0.f, 0.f, 0.f, 0.f};
  v8f oacc[2];
  oacc[0] = vzero; oacc[1] = vzero;

  for (int j = 0; j <= qt; ++j) {       // causal kv-tile loop
#if HAVE_TDM
    if (w == 0) {
      tdm_load_2d(Ktg + (size_t)bh * HD * SEQ + j * 64, Kt, 64, 64, SEQ, 4, 3);
      tdm_load_2d(Vh + ((size_t)bh * SEQ + j * 64) * HD, Vs, 64, 64, HD, 4, 3);
      __builtin_amdgcn_s_wait_tensorcnt(0);
    }
#else
#pragma unroll
    for (int e = 0; e < 2; ++e) {
      int c = e * 256 + tid;
      int r = c >> 3, col = (c & 7) * 8;
      async_copy_b128(Ktg + ((size_t)bh * HD + r) * SEQ + j * 64 + col, Kt + r * 72 + col);
      async_copy_b128(Vh  + ((size_t)bh * SEQ + j * 64 + r) * HD + col, Vs + r * 72 + col);
    }
    wait_async0();
#endif
    __syncthreads();

    // scores = Q Kt  (each wave: 16q x 32k, K-dim 64 = 2 steps)
    v16h aq[2];
    aq[0] = lds_a_frag(Qs, 72, sqr, 0);
    aq[1] = lds_a_frag(Qs, 72, sqr, 32);
    v8f sacc[2];
    sacc[0] = vzero; sacc[1] = vzero;
#pragma unroll
    for (int nt = 0; nt < 2; ++nt)
#pragma unroll
      for (int kc = 0; kc < 2; ++kc) {
        v16h bk = lds_b_frag(Kt, 72, kc * 32, skc + nt * 16);
        sacc[nt] = wmma_f16(aq[kc], bk, sacc[nt]);
      }

    // fp32 epilogue: scale + rel-bias gather + SiLU + causal mask -> LDS f16
#pragma unroll
    for (int nt = 0; nt < 2; ++nt)
#pragma unroll
      for (int r = 0; r < 8; ++r) {
        int lq = sqr + C_ROW(r, lane);
        int lk = skc + nt * 16 + C_COL(lane);
        int qq = q0 + lq;
        int kk = j * 64 + lk;
        float v = sacc[nt][r] * scale +
                  rel_table[(size_t)(qq - kk + MAXSEQ - 1) * HEADS + h];
        v = silu_f(v);
        if (kk > qq) v = 0.f;
        Ss[lq * 72 + lk] = (_Float16)v;
      }
    __syncthreads();

    // out += scores x V  (each wave: 16q x 32d, K-dim 64 = 2 steps)
    v16h as[2];
    as[0] = lds_a_frag(Ss, 72, sqr, 0);
    as[1] = lds_a_frag(Ss, 72, sqr, 32);
#pragma unroll
    for (int nt = 0; nt < 2; ++nt)
#pragma unroll
      for (int kc = 0; kc < 2; ++kc) {
        v16h bv = lds_b_frag(Vs, 72, kc * 32, skc + nt * 16);
        oacc[nt] = wmma_f16(as[kc], bv, oacc[nt]);
      }
    __syncthreads();                    // Kt/Vs/Ss reusable next j
  }

  // gate with U and store f16 "gated" activations
#pragma unroll
  for (int nt = 0; nt < 2; ++nt)
#pragma unroll
    for (int r = 0; r < 8; ++r) {
      int qq = q0 + sqr + C_ROW(r, lane);
      int dd = h * HD + skc + nt * 16 + C_COL(lane);
      size_t m = (size_t)bb * SEQ + qq;
      float v = oacc[nt][r] * (float)Uh[m * HID + dd];
      G[m * HID + dd] = (_Float16)v;
    }
}

// =====================================================================
// Kernel 3: out = gated @ W2h + b2 (fp32 output), double-buffered async.
// =====================================================================
__global__ __launch_bounds__(256)
void k_out(const _Float16* __restrict__ G, const _Float16* __restrict__ W2h,
           const float* __restrict__ b2, float* __restrict__ out) {
  __shared__ __align__(16) _Float16 As[2][128 * 40];
  __shared__ __align__(16) _Float16 Bs[2][32 * 136];

  const int m0 = blockIdx.x * 128;
  const int n0 = blockIdx.y * 128;
  const int tid = threadIdx.x;
  const int w = tid >> 5;
  const unsigned lane = tid & 31u;
  const int wm = (w & 3) * 32;
  const int wn = (w >> 2) * 64;

  const v8f vzero = {0.f, 0.f, 0.f, 0.f, 0.f, 0.f, 0.f, 0.f};
  v8f acc[2][4];
#pragma unroll
  for (int i = 0; i < 2; ++i)
#pragma unroll
    for (int j = 0; j < 4; ++j) acc[i][j] = vzero;

  fill_gemm_tiles(G, HID, W2h, HID, As[0], Bs[0], tid, m0, n0, 0);

  for (int it = 0; it < HID / 32; ++it) {
    wait_async0();
    __syncthreads();
    const int cur = it & 1;
    if (it + 1 < HID / 32)
      fill_gemm_tiles(G, HID, W2h, HID, As[1 - cur], Bs[1 - cur], tid, m0, n0, (it + 1) * 32);

    v16h a[2], b[4];
#pragma unroll
    for (int i = 0; i < 2; ++i) a[i] = lds_a_frag(As[cur], 40, wm + i * 16, 0);
#pragma unroll
    for (int j = 0; j < 4; ++j) b[j] = lds_b_frag(Bs[cur], 136, 0, wn + j * 16);
#pragma unroll
    for (int i = 0; i < 2; ++i)
#pragma unroll
      for (int j = 0; j < 4; ++j)
        acc[i][j] = wmma_f16(a[i], b[j], acc[i][j]);
  }

#pragma unroll
  for (int i = 0; i < 2; ++i)
#pragma unroll
    for (int j = 0; j < 4; ++j)
#pragma unroll
      for (int r = 0; r < 8; ++r) {
        int row = m0 + wm + i * 16 + C_ROW(r, lane);
        int col = n0 + wn + j * 16 + C_COL(lane);
        out[(size_t)row * HID + col] = acc[i][j][r] + b2[col];
      }
}

extern "C" void kernel_launch(void* const* d_in, const int* in_sizes, int n_in,
                              void* d_out, int out_size, void* d_ws, size_t ws_size,
                              hipStream_t stream) {
  (void)in_sizes; (void)n_in; (void)out_size; (void)ws_size;
  const float* x   = (const float*)d_in[0];
  const float* W1  = (const float*)d_in[1];
  const float* b1  = (const float*)d_in[2];
  const float* W2  = (const float*)d_in[3];
  const float* b2  = (const float*)d_in[4];
  const float* rel = (const float*)d_in[5];
  // d_in[6] = causal mask: implemented analytically (tril), not read.

  const size_t NE  = (size_t)BATCH * SEQ * HID;  // 4M elements
  const size_t NW1 = (size_t)HID * FFN;          // 4M
  const size_t NW2 = (size_t)HID * HID;          // 1M
  _Float16* xh  = (_Float16*)d_ws;
  _Float16* W1h = xh  + NE;
  _Float16* W2h = W1h + NW1;
  _Float16* Uh  = W2h + NW2;
  _Float16* Qh  = Uh  + NE;
  _Float16* Ktg = Qh  + NE;                      // K stored [b,h,d,s]
  _Float16* Vh  = Ktg + NE;
  _Float16* G   = Vh  + NE;
  float* out = (float*)d_out;

  dim3 blk(256);
  size_t ncvt = (NE + NW1 + NW2) / 8;
  k_cvt<<<dim3((unsigned)((ncvt + 255) / 256)), blk, 0, stream>>>(x, W1, W2, xh, W1h, W2h);
  k_proj<<<dim3((BATCH * SEQ) / 128, FFN / 128), blk, 0, stream>>>(xh, W1h, b1, Uh, Qh, Ktg, Vh);
  k_attn<<<dim3(BATCH * HEADS, SEQ / 64), blk, 0, stream>>>(Qh, Ktg, Vh, Uh, rel, G);
  k_out<<<dim3((BATCH * SEQ) / 128, HID / 128), blk, 0, stream>>>(G, W2h, b2, out);
}
